// SynthesisLayer_62594853372215
// MI455X (gfx1250) — matrix operand order repaired and checked by reading, not compiled, not run on hardware
//
#include <hip/hip_runtime.h>
#include <hip/hip_bf16.h>
#include <cmath>

#ifndef M_PI
#define M_PI 3.14159265358979323846
#endif

#define B_      8
#define NCH     512
#define INSZ    64
#define HW1     66              // 64 + 3 - 1 (full depthwise conv)
#define N_HW    (HW1*HW1)       // 4356
#define NP      4480            // padded GEMM N (35 * 128)
#define UPH     138             // spatial size after up-filter path
#define PADLO   9

typedef __bf16 bf16s;
typedef bf16s v16bf __attribute__((ext_vector_type(16)));
typedef float v8f   __attribute__((ext_vector_type(8)));

struct F12        { float f[12]; };
struct SirenConst { float grid[72]; float fkg[4]; float gauss[9]; };

// ---------------------------------------------------------------- small prep
__global__ void k_styles(const float* __restrict__ w, const float* __restrict__ aw,
                         const float* __restrict__ ab, float* __restrict__ styles) {
  int idx = blockIdx.x * blockDim.x + threadIdx.x;      // 4096
  int b = idx >> 9, i = idx & 511;
  const float* wr = w + b * 512;
  const float* ar = aw + (size_t)i * 512;
  float acc = 0.f;
  for (int j = 0; j < 512; ++j) acc += wr[j] * ar[j];
  styles[idx] = acc * 0.04419417382415922f + ab[i];     // 1/sqrt(512)
}

__global__ void k_swinv(const float* __restrict__ sw, float* __restrict__ swInv) {
  int o = blockIdx.x * blockDim.x + threadIdx.x;
  if (o >= 512) return;
  const float* r = sw + (size_t)o * 512;
  float s = 0.f;
  for (int i = 0; i < 512; ++i) s += r[i] * r[i];
  swInv[o] = rsqrtf(s * (1.f / 512.f));
}

__global__ void k_sninv(const float* __restrict__ styles, float* __restrict__ snInv) {
  __shared__ float red[256];
  float s = 0.f;
  for (int i = threadIdx.x; i < 4096; i += 256) { float v = styles[i]; s += v * v; }
  red[threadIdx.x] = s; __syncthreads();
  for (int st = 128; st > 0; st >>= 1) {
    if (threadIdx.x < st) red[threadIdx.x] += red[threadIdx.x + st];
    __syncthreads();
  }
  if (threadIdx.x == 0) snInv[0] = rsqrtf(red[0] * (1.f / 4096.f));
}

// one wave per (b,o): compute modulated + demodulated weights, store bf16
__global__ void k_wb(const float* __restrict__ sw, const float* __restrict__ styles,
                     const float* __restrict__ swInv, const float* __restrict__ snInv,
                     __hip_bfloat16* __restrict__ wb) {
  int wave = blockIdx.x * (blockDim.x >> 5) + (threadIdx.x >> 5);   // 0..4095
  int lane = threadIdx.x & 31;
  int b = wave >> 9, o = wave & 511;
  float sI = swInv[o], nI = snInv[0];
  const float* swr = sw + (size_t)o * 512;
  const float* str = styles + (size_t)b * 512;
  float vals[16], ss = 0.f;
  for (int c = 0; c < 16; ++c) {
    int i = c * 32 + lane;
    float v = swr[i] * sI * str[i] * nI;
    vals[c] = v; ss += v * v;
  }
  for (int m = 16; m > 0; m >>= 1) ss += __shfl_xor(ss, m, 32);
  float inv = rsqrtf(ss + 1e-8f);
  __hip_bfloat16* out = wb + ((size_t)wave << 9);
  for (int c = 0; c < 16; ++c) out[c * 32 + lane] = __float2bfloat16(vals[c] * inv);
}

// ------------------------------------------- SIREN -> per-channel 3x3 kernels
__global__ void k_siren(SirenConst sc,
                        const float* __restrict__ w0, const float* __restrict__ b0,
                        const float* __restrict__ w1, const float* __restrict__ b1,
                        const float* __restrict__ w2, const float* __restrict__ b2,
                        const float* __restrict__ w3, const float* __restrict__ b3,
                        const float* __restrict__ ksc, const float* __restrict__ mag,
                        float* __restrict__ kw) {
  __shared__ float hA[36 * 64], hB[36 * 64];
  int t = threadIdx.x;
  for (int e = t; e < 36 * 64; e += 256) {
    int p = e >> 6, j = e & 63;
    float gx = sc.grid[p * 2], gy = sc.grid[p * 2 + 1];
    hA[e] = sinf(30.f * (gx * w0[j] + gy * w0[64 + j] + b0[j]));
  }
  __syncthreads();
  for (int e = t; e < 36 * 64; e += 256) {
    int p = e >> 6, j = e & 63;
    float a = b1[j];
    for (int k = 0; k < 64; ++k) a += hA[p * 64 + k] * w1[k * 64 + j];
    hB[e] = sinf(a);
  }
  __syncthreads();
  for (int e = t; e < 36 * 64; e += 256) {
    int p = e >> 6, j = e & 63;
    float a = b2[j];
    for (int k = 0; k < 64; ++k) a += hB[p * 64 + k] * w2[k * 64 + j];
    hA[e] = sinf(a);
  }
  __syncthreads();
  float invMag = rsqrtf(mag[0]);
  for (int c = t; c < 512; c += 256) {
    float sig = 2.f / (1.f + expf(-ksc[c] * 0.1f));
    float k6[6][6];
    for (int p = 0; p < 36; ++p) {
      float a = b3[c];
      for (int k = 0; k < 64; ++k) a += hA[p * 64 + k] * w3[k * 512 + c];
      k6[p / 6][p % 6] = a * sig;
    }
    // pad 1 each side -> 8x8, 4-tap vertical then horizontal (flipped taps), decimate ::2
    float v[5][8];
    for (int y = 0; y < 5; ++y)
      for (int x = 0; x < 8; ++x) {
        float a = 0.f;
        for (int tt = 0; tt < 4; ++tt) {
          int r = y + tt - 1, cx = x - 1;
          if (r >= 0 && r < 6 && cx >= 0 && cx < 6) a += sc.fkg[tt] * k6[r][cx];
        }
        v[y][x] = a;
      }
    float hh[9];
    for (int yi = 0; yi < 3; ++yi)
      for (int xi = 0; xi < 3; ++xi) {
        float a = 0.f;
        for (int tt = 0; tt < 4; ++tt) a += sc.fkg[tt] * v[yi * 2][xi * 2 + tt];
        hh[yi * 3 + xi] = a;
      }
    float m = 0.f;
    for (int e = 0; e < 9; ++e) m += hh[e];
    m *= (1.f / 9.f);
    float ss = 0.f;
    for (int e = 0; e < 9; ++e) { hh[e] -= m; ss += hh[e] * hh[e]; }
    float inv = rsqrtf(ss * (1.f / 9.f));
    for (int e = 0; e < 9; ++e)
      kw[c * 9 + e] = hh[e] * inv * sc.gauss[e] * invMag;
  }
}

// -------- depthwise 3x3 full conv + pack transposed bf16 Xt[b][n][i] for WMMA B
__global__ void __launch_bounds__(256)
k_dwpack(const float* __restrict__ x, const float* __restrict__ kw,
         __hip_bfloat16* __restrict__ Xt) {
  __shared__ float tile[32][65];
  int b = blockIdx.z, i0 = blockIdx.y * 32, n0 = blockIdx.x * 64;
  int iL = threadIdx.x >> 3, nsg = threadIdx.x & 7;
  int ch = i0 + iL;
  const float* xin = x + ((size_t)(b * NCH + ch)) * (INSZ * INSZ);
  float kv[9];
  for (int e = 0; e < 9; ++e) kv[e] = kw[ch * 9 + e];
  for (int u = 0; u < 8; ++u) {
    int n = n0 + nsg * 8 + u;
    float a = 0.f;
    if (n < N_HW) {
      int y = n / HW1, xx = n - y * HW1;
      #pragma unroll
      for (int dy = 0; dy < 3; ++dy) {
        int sy = y + dy - 2;
        if (sy < 0 || sy >= INSZ) continue;
        #pragma unroll
        for (int dx = 0; dx < 3; ++dx) {
          int sx = xx + dx - 2;
          if (sx < 0 || sx >= INSZ) continue;
          a += kv[dy * 3 + dx] * xin[sy * INSZ + sx];
        }
      }
    }
    tile[iL][nsg * 8 + u] = a;
  }
  __syncthreads();
  int nL = threadIdx.x >> 2, iseg = (threadIdx.x & 3) * 8;
  __hip_bfloat16* op = Xt + ((size_t)b * NP + n0 + nL) * 512 + i0 + iseg;
  for (int u = 0; u < 8; ++u) op[u] = __float2bfloat16(tile[iseg + u][nL]);
}

// --------------------------- modulated 1x1 conv: bf16 WMMA batched GEMM + bias
// A tile (128x32 of wb) staged into double-buffered LDS via the CDNA5 async
// global->LDS path (ASYNCcnt); B fragments read straight from global since the
// packed Xt layout makes them two contiguous b128 loads per fragment.
__global__ void __launch_bounds__(256)
k_gemm(const __hip_bfloat16* __restrict__ wb, const __hip_bfloat16* __restrict__ Xt,
       const float* __restrict__ bias, float* __restrict__ out) {
  __shared__ __align__(16) unsigned short lA[2][128 * 40];  // padded stride 40
  const int b  = blockIdx.z;
  const int Mb = blockIdx.y * 128;
  const int Nb = blockIdx.x * 128;
  const int tid  = threadIdx.x;
  const int wave = tid >> 5;
  const int lane = tid & 31;
  const int mw = wave >> 2;              // 0..1  (64-row slab)
  const int nw = wave & 3;               // 0..3  (32-col slab)
  const int half = lane >> 4;
  const int l16  = lane & 15;

  v8f acc[4][2];
  for (int i = 0; i < 4; ++i)
    for (int j = 0; j < 2; ++j)
      acc[i][j] = v8f{0.f, 0.f, 0.f, 0.f, 0.f, 0.f, 0.f, 0.f};

  const __hip_bfloat16* wbB = wb + ((size_t)b * 512 + Mb) * 512;
  const int am = tid >> 1;               // staged row (0..127)
  const int ak = (tid & 1) * 16;         // staged k offset (0/16)
  // per-thread global source base (advances 64B per K-step) and LDS dests.
  const unsigned long long gbase =
      (unsigned long long)(uintptr_t)(wbB + (size_t)am * 512 + ak);
  const unsigned ldsoff0 = (unsigned)(uintptr_t)(&lA[0][am * 40 + ak]);
  const unsigned ldsoff1 = (unsigned)(uintptr_t)(&lA[1][am * 40 + ak]);

  // stage 32 bytes (two async b128, imm offset applies to both LDS and global)
  #define STAGE_A(kk_)                                                        \
    {                                                                         \
      unsigned long long ga = gbase + (unsigned long long)(kk_) * 64ull;      \
      unsigned lo = ((kk_) & 1) ? ldsoff1 : ldsoff0;                          \
      asm volatile("global_load_async_to_lds_b128 %0, %1, off\n\t"            \
                   "global_load_async_to_lds_b128 %0, %1, off offset:16"      \
                   :: "v"(lo), "v"(ga) : "memory");                           \
    }

  STAGE_A(0);
  for (int kk = 0; kk < 16; ++kk) {
    // fence own async copies, then block barrier => staged tile visible,
    // and everyone is done reading the buffer we are about to overwrite.
    asm volatile("s_wait_asynccnt 0x0" ::: "memory");
    __syncthreads();
    if (kk + 1 < 16) STAGE_A(kk + 1);

    const unsigned short* lAc = &lA[kk & 1][0];

    // B fragments from global: lane col = l16, k = 16*half + e (contiguous)
    union { uint4 u[2]; v16bf v; } bfg[2];
    #pragma unroll
    for (int fn = 0; fn < 2; ++fn) {
      int n = Nb + nw * 32 + fn * 16 + l16;
      const uint4* bp = (const uint4*)(Xt + ((size_t)b * NP + n) * 512 + kk * 32 + half * 16);
      bfg[fn].u[0] = bp[0];
      bfg[fn].u[1] = bp[1];
    }
    #pragma unroll
    for (int fm = 0; fm < 4; ++fm) {
      // A fragment: lane row = l16; k = 16*half + e (contiguous)
      union { uint4 u[2]; v16bf v; } af;
      const int mrow = mw * 64 + fm * 16 + l16;
      af.u[0] = *(const uint4*)(lAc + mrow * 40 + half * 16);
      af.u[1] = *(const uint4*)(lAc + mrow * 40 + half * 16 + 8);
      #pragma unroll
      for (int fn = 0; fn < 2; ++fn) {
        acc[fm][fn] = __builtin_amdgcn_wmma_f32_16x16x32_bf16(
            false, af.v, false, bfg[fn].v, (short)0, acc[fm][fn], false, false);
      }
    }
  }
  #undef STAGE_A

  // epilogue: D lane mapping (col = l16, rows = 8*half + v) + bias
  for (int fm = 0; fm < 4; ++fm)
    for (int fn = 0; fn < 2; ++fn) {
      int col = Nb + nw * 32 + fn * 16 + l16;
      if (col >= N_HW) continue;
      int row0 = Mb + mw * 64 + fm * 16 + half * 8;
      float* op = out + ((size_t)(b * NCH) + row0) * N_HW + col;
      #pragma unroll
      for (int v = 0; v < 8; ++v)
        op[(size_t)v * N_HW] = acc[fm][fn][v] + bias[row0 + v];
    }
}

// --------------------------------------------------- upfirdn up(2) + act path
__global__ void k_vup(const float* __restrict__ e, __hip_bfloat16* __restrict__ t1, F12 fu) {
  size_t idx = (size_t)blockIdx.x * 256 + threadIdx.x;
  const size_t total = (size_t)B_ * NCH * UPH * HW1;
  if (idx >= total) return;
  int xs = (int)(idx % HW1);
  size_t r1 = idx / HW1;
  int r = (int)(r1 % UPH);
  size_t bc = r1 / UPH;
  const float* col = e + bc * N_HW + xs;
  float a = 0.f;
  #pragma unroll
  for (int t = 0; t < 12; ++t) {
    int s = r + t - PADLO;
    if (s >= 0 && (s & 1) == 0) {
      int sr = s >> 1;
      if (sr < HW1) a += fu.f[t] * col[(size_t)sr * HW1];
    }
  }
  t1[idx] = __float2bfloat16(a);
}

__global__ void k_hup_act(const __hip_bfloat16* __restrict__ t1,
                          __hip_bfloat16* __restrict__ tup, F12 fu) {
  size_t idx = (size_t)blockIdx.x * 256 + threadIdx.x;
  const size_t total = (size_t)B_ * NCH * UPH * UPH;
  if (idx >= total) return;
  int cc = (int)(idx % UPH);
  size_t r1 = idx / UPH;
  int r = (int)(r1 % UPH);
  size_t bc = r1 / UPH;
  const __hip_bfloat16* row = t1 + (bc * UPH + r) * HW1;
  float a = 0.f;
  #pragma unroll
  for (int t = 0; t < 12; ++t) {
    int s = cc + t - PADLO;
    if (s >= 0 && (s & 1) == 0) {
      int sr = s >> 1;
      if (sr < HW1) a += fu.f[t] * __bfloat162float(row[sr]);
    }
  }
  float v = a * 4.f;                                  // gain = UP*UP
  v = (v < 0.f ? 0.2f * v : v) * 1.41421356237f;      // leaky * sqrt(2)
  v = fminf(fmaxf(v, -256.f), 256.f);
  tup[idx] = __float2bfloat16(v);
}

// ------------------------------------------------------- upfirdn down(2) path
__global__ void k_vdown(const __hip_bfloat16* __restrict__ tup,
                        __hip_bfloat16* __restrict__ t2, F12 fd) {
  size_t idx = (size_t)blockIdx.x * 256 + threadIdx.x;
  const size_t total = (size_t)B_ * NCH * 64 * UPH;
  if (idx >= total) return;
  int cc = (int)(idx % UPH);
  size_t r1 = idx / UPH;
  int ro = (int)(r1 % 64);
  size_t bc = r1 / 64;
  const __hip_bfloat16* base = tup + bc * (UPH * UPH) + cc;
  float a = 0.f;
  #pragma unroll
  for (int t = 0; t < 12; ++t)
    a += fd.f[t] * __bfloat162float(base[(size_t)(2 * ro + t) * UPH]);
  t2[idx] = __float2bfloat16(a);
}

__global__ void k_hdown(const __hip_bfloat16* __restrict__ t2,
                        float* __restrict__ out, F12 fd) {
  size_t idx = (size_t)blockIdx.x * 256 + threadIdx.x;
  const size_t total = (size_t)B_ * NCH * 64 * 64;
  if (idx >= total) return;
  int xo = (int)(idx % 64);
  size_t r1 = idx / 64;
  int yo = (int)(r1 % 64);
  size_t bc = r1 / 64;
  const __hip_bfloat16* row = t2 + (bc * 64 + yo) * UPH;
  float a = 0.f;
  #pragma unroll
  for (int t = 0; t < 12; ++t)
    a += fd.f[t] * __bfloat162float(row[2 * xo + t]);
  out[idx] = a;
}

// ---------------------------------------------------------------- host consts
static double bessel_i0(double x) {
  double t = x * 0.5, s = 1.0, term = 1.0;
  for (int k = 1; k < 64; ++k) {
    term *= (t * t) / ((double)k * (double)k);
    s += term;
    if (term < 1e-18 * s) break;
  }
  return s;
}

static void firwin_host(int numtaps, double cutoff, double width, double fs, float* out) {
  double nyq = fs * 0.5;
  double atten = 2.285 * (numtaps - 1) * M_PI * (width / nyq) + 7.95;
  double beta;
  if (atten > 50.0)      beta = 0.1102 * (atten - 8.7);
  else if (atten > 21.0) beta = 0.5842 * pow(atten - 21.0, 0.4) + 0.07886 * (atten - 21.0);
  else                   beta = 0.0;
  double h[16], sum = 0.0, c = cutoff / nyq, i0b = bessel_i0(beta);
  for (int n = 0; n < numtaps; ++n) {
    double m = n - (numtaps - 1) / 2.0;
    double x = c * m;
    double sinc = (x == 0.0) ? 1.0 : sin(M_PI * x) / (M_PI * x);
    double r = (numtaps > 1) ? (2.0 * n / (numtaps - 1) - 1.0) : 0.0;
    double arg = 1.0 - r * r; if (arg < 0.0) arg = 0.0;
    double kais = bessel_i0(beta * sqrt(arg)) / i0b;
    h[n] = c * sinc * kais;
    sum += h[n];
  }
  for (int n = 0; n < numtaps; ++n) out[n] = (float)(h[n] / sum);
}

extern "C" void kernel_launch(void* const* d_in, const int* in_sizes, int n_in,
                              void* d_out, int out_size, void* d_ws, size_t ws_size,
                              hipStream_t stream) {
  const float* x    = (const float*)d_in[0];
  const float* w    = (const float*)d_in[1];
  const float* aw   = (const float*)d_in[2];
  const float* ab   = (const float*)d_in[3];
  const float* swt  = (const float*)d_in[4];
  const float* bias = (const float*)d_in[5];
  const float* sw0  = (const float*)d_in[6];
  const float* sb0  = (const float*)d_in[7];
  const float* sw1  = (const float*)d_in[8];
  const float* sb1  = (const float*)d_in[9];
  const float* sw2  = (const float*)d_in[10];
  const float* sb2  = (const float*)d_in[11];
  const float* sw3  = (const float*)d_in[12];
  const float* sb3  = (const float*)d_in[13];
  const float* ksc  = (const float*)d_in[14];
  const float* mag  = (const float*)d_in[15];
  float* outp = (float*)d_out;

  // ---- workspace layout
  char* ws = (char*)d_ws;
  float* styles = (float*)(ws);
  float* swInv  = (float*)(ws + (16 << 10));
  float* snInv  = (float*)(ws + (18 << 10));
  float* kwBuf  = (float*)(ws + (20 << 10));
  const size_t off_wb = 64 << 10;
  __hip_bfloat16* wbB = (__hip_bfloat16*)(ws + off_wb);
  const size_t off_Xt = off_wb + (size_t)B_ * 512 * 512 * 2;          // +4 MB
  __hip_bfloat16* Xt  = (__hip_bfloat16*)(ws + off_Xt);
  const size_t off_E  = off_Xt + (size_t)B_ * NP * 512 * 2;           // +36.7 MB
  float* Ebuf = (float*)(ws + off_E);
  const size_t off_t1 = off_E + (size_t)B_ * NCH * N_HW * 4;          // +71.4 MB
  __hip_bfloat16* t1  = (__hip_bfloat16*)(ws + off_t1);
  const size_t off_tu = off_t1 + (size_t)B_ * NCH * UPH * HW1 * 2;    // +74.6 MB
  __hip_bfloat16* tup = (__hip_bfloat16*)(ws + off_tu);
  __hip_bfloat16* t2  = (__hip_bfloat16*)(ws + off_wb);               // reuse dead region

  // ---- host-computed FIR constants (deterministic, pure CPU math)
  float fu_raw[12], fd_raw[12], kg_raw[4];
  const double SQ2 = 1.4142135623730951;
  firwin_host(12, 32.0, 2.0 * 32.0 * (SQ2 - 1.0), 128.0, fu_raw);
  firwin_host(12, 32.0, 2.0 * 32.0 * (SQ2 - 1.0), 128.0, fd_raw);
  firwin_host(4, 1.5, 2.0 * 1.5 * (SQ2 - 1.0), 3.0 * pow(2.0, 0.1), kg_raw);

  F12 FU, FD;
  for (int t = 0; t < 12; ++t) { FU.f[t] = fu_raw[11 - t]; FD.f[t] = fd_raw[11 - t]; }

  SirenConst SC;
  for (int t = 0; t < 4; ++t) SC.fkg[t] = kg_raw[3 - t];
  {
    double g[3];
    for (int k = 0; k < 3; ++k) g[k] = exp(-(double)((k - 1) * (k - 1))) / sqrt(M_PI);
    for (int i = 0; i < 3; ++i)
      for (int j = 0; j < 3; ++j)
        SC.gauss[i * 3 + j] = (float)(sqrt(g[i] * g[j]) * 2.0);
    double c6[6];
    for (int i = 0; i < 6; ++i) c6[i] = (2.0 * i + 1.0) / 6.0 - 1.0;   // * (0.5*6/3 = 1)
    for (int r = 0; r < 6; ++r)
      for (int c = 0; c < 6; ++c) {
        SC.grid[(r * 6 + c) * 2 + 0] = (float)c6[c];
        SC.grid[(r * 6 + c) * 2 + 1] = (float)c6[r];
      }
  }

  // ---- pipeline
  k_styles<<<16, 256, 0, stream>>>(w, aw, ab, styles);
  k_swinv <<<2, 256, 0, stream>>>(swt, swInv);
  k_sninv <<<1, 256, 0, stream>>>(styles, snInv);
  k_wb    <<<1024, 128, 0, stream>>>(swt, styles, swInv, snInv, wbB);
  k_siren <<<1, 256, 0, stream>>>(SC, sw0, sb0, sw1, sb1, sw2, sb2, sw3, sb3,
                                  ksc, mag, kwBuf);
  k_dwpack<<<dim3(70, 16, 8), 256, 0, stream>>>(x, kwBuf, Xt);
  k_gemm  <<<dim3(35, 4, 8), 256, 0, stream>>>(wbB, Xt, bias, Ebuf);

  {
    size_t tot = (size_t)B_ * NCH * UPH * HW1;
    k_vup<<<(unsigned)((tot + 255) / 256), 256, 0, stream>>>(Ebuf, t1, FU);
  }
  {
    size_t tot = (size_t)B_ * NCH * UPH * UPH;
    k_hup_act<<<(unsigned)((tot + 255) / 256), 256, 0, stream>>>(t1, tup, FU);
  }
  {
    size_t tot = (size_t)B_ * NCH * 64 * UPH;
    k_vdown<<<(unsigned)((tot + 255) / 256), 256, 0, stream>>>(tup, t2, FD);
  }
  {
    size_t tot = (size_t)B_ * NCH * 64 * 64;
    k_hdown<<<(unsigned)((tot + 255) / 256), 256, 0, stream>>>(t2, outp, FD);
  }
}
